// BlockSparseAttention_15908558864464
// MI455X (gfx1250) — matrix operand order repaired and checked by reading, not compiled, not run on hardware
//
#include <hip/hip_runtime.h>
#include <hip/hip_bf16.h>
#include <math.h>

typedef _Float16 half_t;
typedef __attribute__((ext_vector_type(16))) _Float16 v16h;
typedef __attribute__((ext_vector_type(8)))  _Float16 v8h;
typedef __attribute__((ext_vector_type(4)))  _Float16 v4h;
typedef __attribute__((ext_vector_type(8)))  float    v8f;
typedef __attribute__((ext_vector_type(4)))  float    v4f;

#define HID  1024
#define SEQ  2048
#define NBAT 4
#define NH   16
#define DH   64
#define ATT_SCALE 0.125f   // 64^-0.5

// Build an A-operand (16x32 f16) from two 16B chunks per lane:
// lanes 0-15 hold K{0..15}, lanes 16-31 hold K{16..31}, row = lane%16.
static __device__ __forceinline__ v16h make_a(const half_t* lo_p, const half_t* hi_p) {
  v8h lo = *(const v8h*)lo_p;
  v8h hi = *(const v8h*)hi_p;
  v16h r;
#pragma unroll
  for (int i = 0; i < 8; ++i) { r[i] = lo[i]; r[i + 8] = hi[i]; }
  return r;
}

static __device__ __forceinline__ v8f wmma_f16(v16h a, v16h b, v8f c) {
  return __builtin_amdgcn_wmma_f32_16x16x32_f16(false, a, false, b, (short)0, c,
                                                false, false);
}

// ---------------------------------------------------------------- cast f32->f16
__global__ void cast_f32_to_f16_vec4(const float* __restrict__ src,
                                     half_t* __restrict__ dst, int n4) {
  int idx = blockIdx.x * blockDim.x + threadIdx.x;
  int stride = gridDim.x * blockDim.x;
  for (; idx < n4; idx += stride) {
    v4f f = *(const v4f*)(src + 4 * (size_t)idx);
    v4h h;
#pragma unroll
    for (int i = 0; i < 4; ++i) h[i] = (half_t)f[i];
    *(v4h*)(dst + 4 * (size_t)idx) = h;
  }
}

// ------------------------------------------------- Y = A(f16,MxK) @ W^T(f16,NxK) + b
// MODE 0: f16 out [B,H,S,D] (Q,K)   MODE 1: f16 out [B,H,D,S] (V transposed)
// MODE 2: f32 out [M,N]             One wave computes a 16x64 tile.
template <int MODE>
__global__ void __launch_bounds__(128)
gemm_xwt(const half_t* __restrict__ A, const half_t* __restrict__ W,
         const float* __restrict__ bias, void* __restrict__ out) {
  const int lane = threadIdx.x & 31;
  const int wave = __builtin_amdgcn_readfirstlane(threadIdx.x >> 5);
  const int hlf  = lane >> 4;
  const int l16  = lane & 15;
  const int n0 = blockIdx.x * 64;
  const int m0 = blockIdx.y * 64 + wave * 16;

  const half_t* arow = A + (size_t)(m0 + l16) * HID;

  v8f zero = {};
  v8f acc[4];
#pragma unroll
  for (int nc = 0; nc < 4; ++nc) acc[nc] = zero;

  float bval[4];
#pragma unroll
  for (int nc = 0; nc < 4; ++nc) bval[nc] = bias[n0 + nc * 16 + l16];

#pragma unroll 2
  for (int k0 = 0; k0 < HID; k0 += 32) {
    v16h a = make_a(arow + k0 + 16 * hlf, arow + k0 + 16 * hlf + 8);
    __builtin_prefetch(arow + k0 + 32, 0, 3);  // speculative, drop-on-fault
#pragma unroll
    for (int nc = 0; nc < 4; ++nc) {
      // B operand: lane n = col (W row), 16 contiguous f16 at d-offset 16*hlf
      const half_t* wrow = W + (size_t)(n0 + nc * 16 + l16) * HID + k0 + 16 * hlf;
      v16h b = *(const v16h*)wrow;
      acc[nc] = wmma_f16(a, b, acc[nc]);
    }
  }

#pragma unroll
  for (int nc = 0; nc < 4; ++nc) {
#pragma unroll
    for (int r = 0; r < 8; ++r) {
      float v = acc[nc][r] + bval[nc];
      int m = m0 + r + 8 * hlf;       // C layout: row = r + 8*(lane>=16)
      int e = n0 + nc * 16 + l16;     // col = lane%16
      if (MODE == 2) {
        ((float*)out)[(size_t)m * HID + e] = v;
      } else {
        half_t hv = (half_t)v;
        int bb = m >> 11, s = m & (SEQ - 1);
        int hh = e >> 6,  d = e & (DH - 1);
        if (MODE == 0)
          ((half_t*)out)[(((size_t)(bb * NH + hh)) * SEQ + s) * DH + d] = hv;
        else
          ((half_t*)out)[(((size_t)(bb * NH + hh)) * DH + d) * SEQ + s] = hv;
      }
    }
  }
}

// ------------------------------------------------- block-sparse flash attention
// One wave handles 16 query rows of one (b,h). Column blocks of 32.
__global__ void __launch_bounds__(128)
attn_kernel(const half_t* __restrict__ Q, const half_t* __restrict__ K,
            const half_t* __restrict__ VT, half_t* __restrict__ O) {
  __shared__ half_t plds_all[4][16 * 32];
  const int lane = threadIdx.x & 31;
  const int wave = __builtin_amdgcn_readfirstlane(threadIdx.x >> 5);
  const int hlf  = lane >> 4;
  const int l16  = lane & 15;
  const int h = blockIdx.y;
  const int b = blockIdx.z;
  // wave-uniform row-tile base, forced into an SGPR so all tile-loop control is SALU
  const int i0 = __builtin_amdgcn_readfirstlane((blockIdx.x * 4 + wave) * 16);

  const size_t bh = (size_t)(b * NH + h);
  const half_t* qb = Q  + bh * SEQ * DH;
  const half_t* kb = K  + bh * SEQ * DH;
  const half_t* vb = VT + bh * DH * SEQ;   // [DH, SEQ]
  half_t* pl = plds_all[wave];

  // Q as A-operand, K split d = [0,32) and [32,64)
  v16h qa[2];
  {
    const half_t* qrow = qb + (size_t)(i0 + l16) * DH;
#pragma unroll
    for (int kk = 0; kk < 2; ++kk)
      qa[kk] = make_a(qrow + 32 * kk + 16 * hlf, qrow + 32 * kk + 16 * hlf + 8);
  }

  // constant-ones B operand: P @ ones gives per-row sums replicated across lanes
  v16h vones;
#pragma unroll
  for (int i = 0; i < 16; ++i) vones[i] = (half_t)1.0f;

  v8f zero = {};
  v8f acc[4];
#pragma unroll
  for (int nc = 0; nc < 4; ++nc) acc[nc] = zero;
  v8f accS = zero;                    // running softmax denominator per row
  float rmax[8];
#pragma unroll
  for (int r = 0; r < 8; ++r) rmax[r] = -__builtin_inff();

  const float NEG = -__builtin_inff();

  auto process = [&](int j0) {
    // scores for 32 columns = two 16x16 C tiles
    v8f sc[2];
#pragma unroll
    for (int sub = 0; sub < 2; ++sub) {
      v8f c = zero;
      const half_t* krow = kb + (size_t)(j0 + sub * 16 + l16) * DH + 16 * hlf;
      c = wmma_f16(qa[0], *(const v16h*)(krow), c);
      c = wmma_f16(qa[1], *(const v16h*)(krow + 32), c);
      int j = j0 + sub * 16 + l16;
      int jg = (j < 4);                         // per-lane, hoisted over rows
#pragma unroll
      for (int r = 0; r < 8; ++r) {
        int i = i0 + r + 8 * hlf;
        // branchless: |i-j|<=32  <=>  (i-j+32) in [0,64] as unsigned
        int keep = (int)((unsigned)(i - j + 32) <= 64u) | (int)(i < 4) | jg;
        c[r] = keep ? (c[r] * ATT_SCALE) : NEG; // v_cndmask, EXEC untouched
      }
      sc[sub] = c;
    }
    // online softmax: only the row-max needs lane shuffles (sum goes via WMMA)
    float corr[8];
#pragma unroll
    for (int r = 0; r < 8; ++r) {
      float tmax = fmaxf(sc[0][r], sc[1][r]);
#pragma unroll
      for (int m = 1; m < 16; m <<= 1) tmax = fmaxf(tmax, __shfl_xor(tmax, m, 32));
      float nm = fmaxf(rmax[r], tmax);
      float cr = __expf(rmax[r] - nm);
      rmax[r] = nm;
      sc[0][r] = __expf(sc[0][r] - nm);
      sc[1][r] = __expf(sc[1][r] - nm);
      corr[r] = cr;
    }
#pragma unroll
    for (int nc = 0; nc < 4; ++nc)
#pragma unroll
      for (int r = 0; r < 8; ++r) acc[nc][r] *= corr[r];
#pragma unroll
    for (int r = 0; r < 8; ++r) accS[r] *= corr[r];

    // P (16x32 f16) -> LDS, reload in A layout
#pragma unroll
    for (int sub = 0; sub < 2; ++sub)
#pragma unroll
      for (int r = 0; r < 8; ++r)
        pl[(r + 8 * hlf) * 32 + sub * 16 + l16] = (half_t)sc[sub][r];
    asm volatile("s_wait_dscnt 0" ::: "memory");
    const half_t* pr = pl + l16 * 32;
    v16h pa = make_a(pr + 16 * hlf, pr + 16 * hlf + 8);

    // denominator: row-sum of P via matrix unit (replicated over the 16-lane half)
    accS = wmma_f16(pa, vones, accS);

    // P @ V : V^T rows are contiguous B-operand lanes (lane n = d, 16 j's)
#pragma unroll
    for (int nc = 0; nc < 4; ++nc) {
      const half_t* vrow = vb + (size_t)(nc * 16 + l16) * SEQ + j0 + 16 * hlf;
      acc[nc] = wmma_f16(pa, *(const v16h*)vrow, acc[nc]);
    }
  };

  if (i0 == 0) {
    for (int j0 = 0; j0 < SEQ; j0 += 32) process(j0);   // global rows need full sweep
  } else {
    process(0);                                          // global columns block
    int wb0 = (i0 - 32) >> 5;                            // 3 window blocks
    for (int t = 0; t < 3; ++t) {
      int wb = wb0 + t;
      if (wb >= 1 && wb < SEQ / 32) process(wb * 32);
    }
  }

  float inv[8];
#pragma unroll
  for (int r = 0; r < 8; ++r) inv[r] = 1.0f / accS[r];
#pragma unroll
  for (int nc = 0; nc < 4; ++nc) {
#pragma unroll
    for (int r = 0; r < 8; ++r) {
      int s = i0 + r + 8 * hlf;
      int d = nc * 16 + l16;
      O[((size_t)b * SEQ + s) * HID + h * DH + d] = (half_t)(acc[nc][r] * inv[r]);
    }
  }
}

// ---------------------------------------------------------------- launch
extern "C" void kernel_launch(void* const* d_in, const int* in_sizes, int n_in,
                              void* d_out, int out_size, void* d_ws, size_t ws_size,
                              hipStream_t stream) {
  (void)in_sizes; (void)n_in; (void)out_size; (void)ws_size;
  const float* x  = (const float*)d_in[0];
  const float* Wq = (const float*)d_in[1];
  const float* bq = (const float*)d_in[2];
  const float* Wk = (const float*)d_in[3];
  const float* bk = (const float*)d_in[4];
  const float* Wv = (const float*)d_in[5];
  const float* bv = (const float*)d_in[6];
  const float* Wo = (const float*)d_in[7];
  const float* bo = (const float*)d_in[8];

  const size_t NX = (size_t)NBAT * SEQ * HID;  // 8388608
  const size_t NW = (size_t)HID * HID;         // 1048576

  half_t* xh  = (half_t*)d_ws;
  half_t* wqh = xh  + NX;
  half_t* wkh = wqh + NW;
  half_t* wvh = wkh + NW;
  half_t* woh = wvh + NW;
  half_t* qh  = woh + NW;   // [B,H,S,D]
  half_t* kh  = qh  + NX;   // [B,H,S,D]
  half_t* vth = kh  + NX;   // [B,H,D,S]
  half_t* aoh = vth + NX;   // [B,S,HID]

  cast_f32_to_f16_vec4<<<4096, 256, 0, stream>>>(x,  xh,  (int)(NX / 4));
  cast_f32_to_f16_vec4<<<1024, 256, 0, stream>>>(Wq, wqh, (int)(NW / 4));
  cast_f32_to_f16_vec4<<<1024, 256, 0, stream>>>(Wk, wkh, (int)(NW / 4));
  cast_f32_to_f16_vec4<<<1024, 256, 0, stream>>>(Wv, wvh, (int)(NW / 4));
  cast_f32_to_f16_vec4<<<1024, 256, 0, stream>>>(Wo, woh, (int)(NW / 4));

  dim3 ggrid(HID / 64, (NBAT * SEQ) / 64);  // (16, 128)
  gemm_xwt<0><<<ggrid, 128, 0, stream>>>(xh, wqh, bq, qh);
  gemm_xwt<0><<<ggrid, 128, 0, stream>>>(xh, wkh, bk, kh);
  gemm_xwt<1><<<ggrid, 128, 0, stream>>>(xh, wvh, bv, vth);

  attn_kernel<<<dim3(SEQ / 64, NH, NBAT), 128, 0, stream>>>(qh, kh, vth, aoh);

  gemm_xwt<2><<<ggrid, 128, 0, stream>>>(aoh, woh, bo, d_out);
}
